// CoarseModule_54245436948733
// MI455X (gfx1250) — compile-verified, matrix operand order checked by reading, not run on hardware
//
#include <hip/hip_runtime.h>
#include <hip/hip_bf16.h>

#define HW   2304      // 48*48 spatial positions (Q and S)
#define CDIM 256       // channels (K)
#define NB   8         // batch
#define LDK  40        // padded LDS K stride (halfs)

typedef __attribute__((ext_vector_type(16))) _Float16 v16h;
typedef __attribute__((ext_vector_type(8)))  _Float16 v8h;
typedef __attribute__((ext_vector_type(8)))  float    v8f;
typedef __attribute__((ext_vector_type(4)))  float    v4f;

// ---- WMMA fragment load from an LDS tile stored [row][k] (row stride LDK halfs).
// 16-bit A/B 16x32 layout (ISA 7.12.2): lanes 0-15 -> K {0..7, 16..23},
// lanes 16-31 -> K {8..15, 24..31}; row index = (lane & 15) within subtile.
__device__ __forceinline__ v16h ldsFrag(const _Float16* base, int row, int lane) {
  int kOff = (lane >> 4) << 3;                  // 0 or 8
  const _Float16* p = base + row * LDK + kOff;  // 16B aligned (LDK*2=80, kOff*2 in {0,16})
  v8h lo = *(const v8h*)p;                      // K = kOff .. kOff+7
  v8h hi = *(const v8h*)(p + 16);               // K = kOff+16 .. kOff+23
  return __builtin_shufflevector(lo, hi, 0,1,2,3,4,5,6,7,8,9,10,11,12,13,14,15);
}

// Patch-view row permutation: q=(h*4+ph)*48+(w*4+pw) -> (h*12+w)*16 + ph*4+pw
__device__ __forceinline__ int perm16(int q) {
  int row = q / 48, col = q - row * 48;
  return (((row >> 2) * 12 + (col >> 2)) << 4) + ((row & 3) << 2) + (col & 3);
}

// ---------------------------------------------------------------------------
// K1: sim[n,q,s] = (10/256) * sum_c Qf[n,c,q] * Mf[n,c,s]
// Workgroup tile 128(q) x 64(s), 8 waves; split-f16 (hi+lo) -> 3 WMMAs/subtile.
// ---------------------------------------------------------------------------
__global__ __launch_bounds__(256)
void gemm_sim(const float* __restrict__ Qf, const float* __restrict__ Mf,
              float* __restrict__ sim) {
  __shared__ __align__(16) _Float16 Ah[128 * LDK];
  __shared__ __align__(16) _Float16 Al[128 * LDK];
  __shared__ __align__(16) _Float16 Bh[64 * LDK];
  __shared__ __align__(16) _Float16 Bl[64 * LDK];

  const int tid  = threadIdx.x;
  const int lane = tid & 31, wave = tid >> 5;
  const int n     = blockIdx.z;
  const int qBase = blockIdx.y * 128;
  const int sBase = blockIdx.x * 64;
  const float* Ag = Qf + (size_t)n * CDIM * HW;   // [c][q], q contiguous
  const float* Bg = Mf + (size_t)n * CDIM * HW;   // [c][s], s contiguous

  v8f acc[4] = {};

  for (int k0 = 0; k0 < CDIM; k0 += 32) {
    __syncthreads();
    // Stage A: 32(k) x 128(q) fp32 -> transposed f16 hi/lo tiles [q][k]
#pragma unroll
    for (int i = 0; i < 4; ++i) {
      int linear = tid + i * 256;            // 0..1023
      int cl = linear >> 5, q4 = linear & 31;
      v4f v = *(const v4f*)(Ag + (size_t)(k0 + cl) * HW + qBase + q4 * 4);
#pragma unroll
      for (int j = 0; j < 4; ++j) {
        float f = v[j];
        _Float16 h = (_Float16)f;
        Ah[(q4 * 4 + j) * LDK + cl] = h;
        Al[(q4 * 4 + j) * LDK + cl] = (_Float16)(f - (float)h);
      }
    }
    // Stage B: 32(k) x 64(s)
#pragma unroll
    for (int i = 0; i < 2; ++i) {
      int linear = tid + i * 256;            // 0..511
      int cl = linear >> 4, s4 = linear & 15;
      v4f v = *(const v4f*)(Bg + (size_t)(k0 + cl) * HW + sBase + s4 * 4);
#pragma unroll
      for (int j = 0; j < 4; ++j) {
        float f = v[j];
        _Float16 h = (_Float16)f;
        Bh[(s4 * 4 + j) * LDK + cl] = h;
        Bl[(s4 * 4 + j) * LDK + cl] = (_Float16)(f - (float)h);
      }
    }
    __syncthreads();

    int mrow = wave * 16 + (lane & 15);
    v16h ah = ldsFrag(Ah, mrow, lane);
    v16h al = ldsFrag(Al, mrow, lane);
#pragma unroll
    for (int j = 0; j < 4; ++j) {
      int nrow = j * 16 + (lane & 15);
      v16h bh = ldsFrag(Bh, nrow, lane);
      v16h bl = ldsFrag(Bl, nrow, lane);
      acc[j] = __builtin_amdgcn_wmma_f32_16x16x32_f16(false, ah, false, bh, (short)0, acc[j], false, false);
      acc[j] = __builtin_amdgcn_wmma_f32_16x16x32_f16(false, ah, false, bl, (short)0, acc[j], false, false);
      acc[j] = __builtin_amdgcn_wmma_f32_16x16x32_f16(false, al, false, bh, (short)0, acc[j], false, false);
    }
  }

  // C/D layout: lanes 0-15 -> M=r, N=lane; lanes 16-31 -> M=r+8, N=lane-16
  const float scale = 10.0f / 256.0f;
  const int ncol = lane & 15;
  const int mAdd = (lane >> 4) * 8;
#pragma unroll
  for (int j = 0; j < 4; ++j) {
#pragma unroll
    for (int r = 0; r < 8; ++r) {
      int q = qBase + wave * 16 + mAdd + r;
      int s = sBase + j * 16 + ncol;
      sim[((size_t)n * HW + q) * HW + s] = acc[j][r] * scale;
    }
  }
}

// ---------------------------------------------------------------------------
// K2: per-row (fixed n,q) max & sum(exp) over s  -> softmax axis=2 stats
// ---------------------------------------------------------------------------
__global__ __launch_bounds__(256)
void row_stats(const float* __restrict__ sim, float* __restrict__ rmax,
               float* __restrict__ rsum) {
  const int r = blockIdx.x;                 // r = n*HW + q
  const int tid = threadIdx.x;
  const float* row = sim + (size_t)r * HW;
  float v[9];
  float m = -3.0e38f;
#pragma unroll
  for (int k = 0; k < 9; ++k) { v[k] = row[tid + k * 256]; m = fmaxf(m, v[k]); }
  __shared__ float red[256];
  red[tid] = m; __syncthreads();
  for (int off = 128; off > 0; off >>= 1) {
    if (tid < off) red[tid] = fmaxf(red[tid], red[tid + off]);
    __syncthreads();
  }
  m = red[0]; __syncthreads();
  float s = 0.0f;
#pragma unroll
  for (int k = 0; k < 9; ++k) s += __expf(v[k] - m);
  red[tid] = s; __syncthreads();
  for (int off = 128; off > 0; off >>= 1) {
    if (tid < off) red[tid] += red[tid + off];
    __syncthreads();
  }
  if (tid == 0) { rmax[r] = m; rsum[r] = red[0]; }
}

// ---------------------------------------------------------------------------
// K3: per-column (fixed n,s) online softmax stats over q  -> softmax axis=1
// ---------------------------------------------------------------------------
__global__ __launch_bounds__(256)
void col_stats(const float* __restrict__ sim, float* __restrict__ cmax,
               float* __restrict__ csum) {
  const int n = blockIdx.y;
  const int s = blockIdx.x * 256 + threadIdx.x;
  const float* base = sim + (size_t)n * HW * HW + s;
  float m = -3.0e38f, l = 0.0f;
  for (int q = 0; q < HW; ++q) {
    float x = base[(size_t)q * HW];
    float mn = fmaxf(m, x);
    l = l * __expf(m - mn) + __expf(x - mn);
    m = mn;
  }
  cmax[n * HW + s] = m;
  csum[n * HW + s] = l;
}

// ---------------------------------------------------------------------------
// K4: emit instance_masks (+ permuted copy) and query_masks (+ permuted copy)
//     via 32x32 LDS tile transpose so all global writes stay coalesced.
// ---------------------------------------------------------------------------
__global__ __launch_bounds__(256)
void emit_masks(const float* __restrict__ sim,
                const float* __restrict__ rmax, const float* __restrict__ rsum,
                const float* __restrict__ cmax, const float* __restrict__ csum,
                float* __restrict__ inst, float* __restrict__ qmask,
                float* __restrict__ tmpi, float* __restrict__ tmpq) {
  const int n  = blockIdx.z;
  const int qB = blockIdx.y * 32, sB = blockIdx.x * 32;
  const int tid = threadIdx.x;
  const int sl = tid & 31, qr = tid >> 5;
  __shared__ float tile[32][33];
  const float* rm = rmax + n * HW; const float* rs = rsum + n * HW;
  const float* cm = cmax + n * HW; const float* cs = csum + n * HW;
#pragma unroll
  for (int i = 0; i < 4; ++i) {
    int ql = i * 8 + qr;
    int q = qB + ql, s = sB + sl;
    float x = sim[((size_t)n * HW + q) * HW + s];
    float iv = __expf(x - cm[s]) / cs[s];                 // softmax over q (axis=1)
    inst[((size_t)n * HW + q) * HW + s] = iv;
    tmpi[((size_t)n * HW + perm16(q)) * HW + s] = iv;     // (n,144,16,S) view
    tile[sl][ql] = __expf(x - rm[q]) / rs[q];             // softmax over s (axis=2)
  }
  __syncthreads();
#pragma unroll
  for (int i = 0; i < 4; ++i) {
    int s2 = i * 8 + qr, q2 = sl;
    float qv = tile[s2][q2];
    int s = sB + s2, q = qB + q2;
    qmask[((size_t)n * HW + s) * HW + q] = qv;            // (n,S,Q)
    tmpq[((size_t)n * HW + perm16(s)) * HW + q] = qv;     // (n,144,16,Q) view
  }
}

// ---------------------------------------------------------------------------
// K5: patch sums of the (already permuted) tmp buffers + EPS clamp
// ---------------------------------------------------------------------------
__global__ __launch_bounds__(256)
void pool_sum(const float* __restrict__ src, float* __restrict__ dst) {
  const int n = blockIdx.z, p = blockIdx.y;
  const int s = blockIdx.x * 256 + threadIdx.x;
  const float* b = src + ((size_t)n * HW + p * 16) * HW + s;
  float sum = 0.0f;
#pragma unroll
  for (int w = 0; w < 16; ++w) sum += b[(size_t)w * HW];
  dst[((size_t)n * 144 + p) * HW + s] = fminf(sum, 1.0f - 1e-6f);
}

// ---------------------------------------------------------------------------
extern "C" void kernel_launch(void* const* d_in, const int* in_sizes, int n_in,
                              void* d_out, int out_size, void* d_ws, size_t ws_size,
                              hipStream_t stream) {
  (void)in_sizes; (void)n_in; (void)out_size; (void)ws_size;
  const float* mask_feats  = (const float*)d_in[0];   // (N,C,H,W) -> "s" side
  const float* query_feats = (const float*)d_in[1];   // (N,C,H,W) -> "q" side
  // d_in[2] = path_size (==4), baked into perm16/pooling.

  const size_t SZ_MASK = (size_t)NB * HW * HW;   // 42,467,328
  const size_t SZ_D    = (size_t)NB * 144 * HW;  //  2,654,208
  float* out0 = (float*)d_out;            // instance_masks      (N,Q,S)
  float* out1 = out0 + SZ_MASK;           // query_masks         (N,S,Q)
  float* out2 = out1 + SZ_MASK;           // instance_masks_d    (N,144,S)
  float* out3 = out2 + SZ_D;              // query_masks_d       (N,144,Q)
  float* out4 = out3 + SZ_D;              // tmp_instance_masks_d(N,144,16,S)
  float* out5 = out4 + SZ_MASK;           // tmp_query_masks_d   (N,144,16,Q)

  float* sim  = (float*)d_ws;             // 170 MB scratch
  float* rmax = sim + SZ_MASK;
  float* rsum = rmax + NB * HW;
  float* cmax = rsum + NB * HW;
  float* csum = cmax + NB * HW;

  gemm_sim<<<dim3(HW / 64, HW / 128, NB), 256, 0, stream>>>(query_feats, mask_feats, sim);
  row_stats<<<dim3(NB * HW), 256, 0, stream>>>(sim, rmax, rsum);
  col_stats<<<dim3(HW / 256, NB), 256, 0, stream>>>(sim, cmax, csum);
  emit_masks<<<dim3(HW / 32, HW / 32, NB), 256, 0, stream>>>(
      sim, rmax, rsum, cmax, csum, out0, out1, out4, out5);
  pool_sum<<<dim3(HW / 256, 144, NB), 256, 0, stream>>>(out4, out2);
  pool_sum<<<dim3(HW / 256, 144, NB), 256, 0, stream>>>(out5, out3);
}
